// HMM_50706383897287
// MI455X (gfx1250) — matrix operand order repaired and checked by reading, not compiled, not run on hardware
//
#include <hip/hip_runtime.h>
#include <hip/hip_bf16.h>

#define S 1024
#define V 512
#define B 64
#define T 128

typedef __attribute__((ext_vector_type(16))) _Float16 v16h;
typedef __attribute__((ext_vector_type(8)))  _Float16 v8h;
typedef __attribute__((ext_vector_type(8)))  float    v8f;

// Order-preserving float -> uint encoding so we can atomicMax float maxima.
__device__ __forceinline__ unsigned enc_f32(float f) {
    unsigned u = __float_as_uint(f);
    return (u & 0x80000000u) ? ~u : (u | 0x80000000u);
}
__device__ __forceinline__ float dec_f32(unsigned k) {
    return (k & 0x80000000u) ? __uint_as_float(k & 0x7fffffffu)
                             : __uint_as_float(~k);
}

// Row-wise logsumexp: out[row] = LSE(src[row, 0:ncols]); grid = nrows, block = 256.
__global__ void row_lse_kernel(const float* __restrict__ src,
                               float* __restrict__ out, int ncols) {
    __shared__ float red[256];
    const float* p = src + (size_t)blockIdx.x * ncols;
    float m = -INFINITY;
    for (int c = threadIdx.x; c < ncols; c += 256) m = fmaxf(m, p[c]);
    red[threadIdx.x] = m; __syncthreads();
    for (int s = 128; s > 0; s >>= 1) {
        if (threadIdx.x < s) red[threadIdx.x] = fmaxf(red[threadIdx.x], red[threadIdx.x + s]);
        __syncthreads();
    }
    m = red[0]; __syncthreads();
    float sum = 0.f;
    for (int c = threadIdx.x; c < ncols; c += 256) sum += __expf(p[c] - m);
    red[threadIdx.x] = sum; __syncthreads();
    for (int s = 128; s > 0; s >>= 1) {
        if (threadIdx.x < s) red[threadIdx.x] += red[threadIdx.x + s];
        __syncthreads();
    }
    if (threadIdx.x == 0) out[blockIdx.x] = m + __logf(red[0]);
}

__global__ void reset_ckey_kernel(unsigned* __restrict__ cKey, int n) {
    int i = blockIdx.x * 256 + threadIdx.x;
    if (i < n) cKey[i] = 0u;   // 0 < enc(-inf): any real value wins
}

// expT[i,j] = exp(trans[i,j] - rowLSE[i] - colmax[j]) as f16 in a WMMA-B tiled
// layout: element (k=i, n=j) at [ (j/16)*S + i ]*16 + (j%16). grid = S/256.
__global__ void build_expT_kernel(const float* __restrict__ trans,
                                  const float* __restrict__ rlse,
                                  float* __restrict__ colmax,
                                  _Float16* __restrict__ expT) {
    int j = blockIdx.x * 256 + threadIdx.x;
    float m = -INFINITY;
    for (int i = 0; i < S; ++i)
        m = fmaxf(m, trans[(size_t)i * S + j] - rlse[i]);
    colmax[j] = m;
    size_t tbase = ((size_t)(j >> 4) * S) * 16 + (j & 15);
    for (int i = 0; i < S; ++i) {
        float v = trans[(size_t)i * S + j] - rlse[i] - m;
        expT[tbase + (size_t)i * 16] = (_Float16)__expf(v);
    }
}

// alpha0[b,j] = emis_ln[j, value[b,0]] + prior_ln[j]; also c[1][b] = max_j alpha0.
__global__ void init_alpha_kernel(const int* __restrict__ value,
                                  const float* __restrict__ prior,
                                  const float* __restrict__ emis,
                                  const float* __restrict__ priorLSE,
                                  const float* __restrict__ emisLSE,
                                  float* __restrict__ alpha,
                                  unsigned* __restrict__ cKey1) {
    __shared__ float red[256];
    int b = blockIdx.x;
    int v0 = value[b * T];
    float pl = priorLSE[0];
    float m = -INFINITY;
    for (int j = threadIdx.x; j < S; j += 256) {
        float a = (emis[(size_t)j * V + v0] - emisLSE[j]) + (prior[j] - pl);
        alpha[(size_t)b * S + j] = a;
        m = fmaxf(m, a);
    }
    red[threadIdx.x] = m; __syncthreads();
    for (int s = 128; s > 0; s >>= 1) {
        if (threadIdx.x < s) red[threadIdx.x] = fmaxf(red[threadIdx.x], red[threadIdx.x + s]);
        __syncthreads();
    }
    if (threadIdx.x == 0) cKey1[b] = enc_f32(red[0]);
}

// P[b,i] = (f16) exp(alpha[b,i] - c_b): computed once per step (not per tile).
__global__ void make_P_kernel(const float* __restrict__ alpha,
                              const unsigned* __restrict__ cKey,
                              _Float16* __restrict__ P) {
    int b = blockIdx.x;
    float cb = dec_f32(cKey[b]);
    for (int j = threadIdx.x; j < S; j += 256)
        P[(size_t)b * S + j] = (_Float16)__expf(alpha[(size_t)b * S + j] - cb);
}

// One scan step: alphaOut[b,j] = e_t[b,j] + c_b + colmax[j] + log( (P x expT)[b,j] ).
// grid = S/16 blocks x 128 threads (4 waves); wave w owns M-tile w, block owns 16 cols.
__global__ void __launch_bounds__(128)
hmm_step_kernel(const _Float16* __restrict__ P,
                const _Float16* __restrict__ expT,
                const float* __restrict__ colmax,
                const unsigned* __restrict__ cKeyIn,
                unsigned* __restrict__ cKeyOut,
                const float* __restrict__ emis,
                const float* __restrict__ emisLSE,
                const int* __restrict__ value,
                float* __restrict__ alphaOut,
                int t) {
    const int lane = threadIdx.x & 31;
    const int wave = threadIdx.x >> 5;        // 0..3 -> M-tile
    const int n0   = blockIdx.x * 16;         // 16 output columns per block
    const int ml   = lane & 15;
    const int d    = (lane & 16) ? 8 : 0;     // K half-offset per ISA A-layout

    const _Float16* prow  = P + (size_t)(wave * 16 + ml) * S;
    const _Float16* bbase = expT + ((size_t)(n0 >> 4) * S) * 16 + (size_t)ml;

    v8f acc = {};
#pragma unroll 4
    for (int k0 = 0; k0 < S; k0 += 32) {
        // A fragment (16-bit A 16x32 layout): elems 0-7 = K[d..d+7], 8-15 = K[16+d..16+d+7]
        v8h alo = *(const v8h*)(prow + k0 + d);
        v8h ahi = *(const v8h*)(prow + k0 + 16 + d);
        v16h a = __builtin_shufflevector(alo, ahi,
                                         0, 1, 2, 3, 4, 5, 6, 7,
                                         8, 9, 10, 11, 12, 13, 14, 15);
        // B fragment: lane <-> N column, elems 0-7 = K[d..d+7], 8-15 = K[16+d..16+d+7] (stride 16).
        v16h bfrag;
#pragma unroll
        for (int s = 0; s < 16; ++s)
            bfrag[s] = bbase[(size_t)(k0 + (s < 8 ? d + s : 16 + d + (s - 8))) * 16];
        if (k0 + 32 < S)
            __builtin_prefetch(bbase + (size_t)(k0 + 32) * 16, 0, 1);
        acc = __builtin_amdgcn_wmma_f32_16x16x32_f16(
            false, a, false, bfrag, (short)0, acc, false, false);
    }

    const int n   = lane & 15;
    const int j   = n0 + n;
    const float cm_j = colmax[j];
    const float el_j = emisLSE[j];

#pragma unroll
    for (int r = 0; r < 8; ++r) {
        // C/D layout: lanes 0-15 -> M = r, lanes 16-31 -> M = r + 8; N = lane&15.
        int bg   = wave * 16 + r + ((lane & 16) ? 8 : 0);
        float cb = dec_f32(cKeyIn[bg]);
        int   vt = value[bg * T + t];
        float e  = emis[(size_t)j * V + vt] - el_j;
        float out = e + cb + cm_j + __logf(acc[r]);
        alphaOut[(size_t)bg * S + j] = out;
        // Reduce max over the 16 columns in this lane-half, then one atomic.
        float mv = out;
        for (int off = 1; off < 16; off <<= 1)
            mv = fmaxf(mv, __shfl_xor(mv, off, 32));
        if (n == 0) atomicMax(&cKeyOut[bg], enc_f32(mv));
    }
}

__global__ void finalize_kernel(const float* __restrict__ alpha,
                                float* __restrict__ out) {
    __shared__ float red[256];
    int b = blockIdx.x;
    const float* p = alpha + (size_t)b * S;
    float m = -INFINITY;
    for (int j = threadIdx.x; j < S; j += 256) m = fmaxf(m, p[j]);
    red[threadIdx.x] = m; __syncthreads();
    for (int s = 128; s > 0; s >>= 1) {
        if (threadIdx.x < s) red[threadIdx.x] = fmaxf(red[threadIdx.x], red[threadIdx.x + s]);
        __syncthreads();
    }
    m = red[0]; __syncthreads();
    float sum = 0.f;
    for (int j = threadIdx.x; j < S; j += 256) sum += __expf(p[j] - m);
    red[threadIdx.x] = sum; __syncthreads();
    for (int s = 128; s > 0; s >>= 1) {
        if (threadIdx.x < s) red[threadIdx.x] += red[threadIdx.x + s];
        __syncthreads();
    }
    if (threadIdx.x == 0) out[b] = m + __logf(red[0]);
}

extern "C" void kernel_launch(void* const* d_in, const int* in_sizes, int n_in,
                              void* d_out, int out_size, void* d_ws, size_t ws_size,
                              hipStream_t stream) {
    (void)in_sizes; (void)n_in; (void)out_size; (void)ws_size;
    const int*   value = (const int*)d_in[0];
    const float* prior = (const float*)d_in[1];
    const float* trans = (const float*)d_in[2];
    const float* emis  = (const float*)d_in[3];
    float* out = (float*)d_out;

    // Workspace layout (~2.8 MB total)
    char* ws = (char*)d_ws;
    float*    transLSE = (float*)(ws + 0);                         // 4 KB
    float*    emisLSE  = (float*)(ws + 4096);                      // 4 KB
    float*    priorLSE = (float*)(ws + 8192);                      // 64 B
    float*    colmax   = (float*)(ws + 8256);                      // 4 KB
    unsigned* cKey     = (unsigned*)(ws + 12352);                  // 129*64*4 = 33 KB
    _Float16* expT     = (_Float16*)(ws + 45568);                  // 2 MB, 256B aligned
    _Float16* P        = (_Float16*)(ws + 45568 + 2097152);        // 128 KB
    float*    alphaA   = (float*)(ws + 45568 + 2097152 + 131072);  // 256 KB
    float*    alphaB   = alphaA + B * S;                           // 256 KB

    // Preprocessing (all time-invariant; expT stays resident in L2 all run)
    reset_ckey_kernel<<<(129 * 64 + 255) / 256, 256, 0, stream>>>(cKey, 129 * 64);
    row_lse_kernel<<<S, 256, 0, stream>>>(trans, transLSE, S);
    row_lse_kernel<<<S, 256, 0, stream>>>(emis, emisLSE, V);
    row_lse_kernel<<<1, 256, 0, stream>>>(prior, priorLSE, S);
    build_expT_kernel<<<S / 256, 256, 0, stream>>>(trans, transLSE, colmax, expT);
    init_alpha_kernel<<<B, 256, 0, stream>>>(value, prior, emis, priorLSE, emisLSE,
                                             alphaA, cKey + 64);

    // Sequential scan: kernel boundary = global sync per step.
    float* aIn = alphaA; float* aOut = alphaB;
    for (int t = 1; t < T; ++t) {
        make_P_kernel<<<B, 256, 0, stream>>>(aIn, cKey + t * 64, P);
        hmm_step_kernel<<<S / 16, 128, 0, stream>>>(P, expT, colmax,
                                                    cKey + t * 64, cKey + (t + 1) * 64,
                                                    emis, emisLSE, value, aOut, t);
        float* tmp = aIn; aIn = aOut; aOut = tmp;
    }
    finalize_kernel<<<B, 256, 0, stream>>>(aIn, out);
}
